// GlobalAttention_20976620273939
// MI455X (gfx1250) — matrix operand order repaired and checked by reading, not compiled, not run on hardware
//
#include <hip/hip_runtime.h>
#include <stdint.h>

typedef __attribute__((ext_vector_type(16))) __bf16 v16bf;
typedef __attribute__((ext_vector_type(8)))  float  v8f;
typedef int v4i __attribute__((vector_size(16)));

#define N_POS 4096
#define C_DIM 256
#define QK    32
#define B_SZ  4
#define JT    64        // key tile
#define NTILE (N_POS / JT)

static __device__ __forceinline__ unsigned short f2bf(float f) {
    union { float f; unsigned int u; } v; v.f = f;
    unsigned int r = v.u + 0x7FFFu + ((v.u >> 16) & 1u);   // round-nearest-even
    return (unsigned short)(r >> 16);
}

union FragBF { v16bf v; unsigned short u[16]; uint4 q[2]; };
union AccF   { v8f v; float f[8]; float4 q[2]; };

// --------- async global->LDS staging (CDNA5), with compile-safe fallback ----
#if __has_builtin(__builtin_amdgcn_global_load_async_to_lds_b128) && \
    __has_builtin(__builtin_amdgcn_s_wait_asynccnt)
#define ASYNC_LDS 1
#else
#define ASYNC_LDS 0
#endif

static __device__ __forceinline__ void cp16_g2l(unsigned short* l, const unsigned short* g) {
#if ASYNC_LDS
    __builtin_amdgcn_global_load_async_to_lds_b128(
        (__attribute__((address_space(1))) v4i*)(unsigned short*)(g),
        (__attribute__((address_space(3))) v4i*)(l), 0, 0);
#else
    *(uint4*)l = *(const uint4*)g;
#endif
}
static __device__ __forceinline__ void wait_fill() {
#if ASYNC_LDS
    __builtin_amdgcn_s_wait_asynccnt(0);
#endif
}

// ---------------------------------------------------------------------------
// Projection GEMM: out = X^T (N x 256) @ W^T (256 x O) + bias, bf16 in/out.
// storeNO=1 -> (B, N, O)  (Q^T, K^T);  storeNO=0 -> (B, O, N)  (V)
// ---------------------------------------------------------------------------
__global__ void __launch_bounds__(128) proj_kernel(
    const float* __restrict__ x, const float* __restrict__ w,
    const float* __restrict__ bias, unsigned short* __restrict__ out,
    int O, int storeNO)
{
    const int lane = threadIdx.x & 31;
    const int wid  = threadIdx.x >> 5;
    const int n    = lane & 15;
    const int half = lane >> 4;
    const int b    = blockIdx.z;
    const int posBase = blockIdx.x * 64 + wid * 16;
    const int colBase = blockIdx.y * 16;

    const float* xb = x + b * (C_DIM * N_POS);
    v8f acc = {};

    for (int kb = 0; kb < C_DIM; kb += 32) {
        FragBF af;                                   // A[m=n][k], K-map runs
        #pragma unroll
        for (int e = 0; e < 16; ++e) {
            int K = half * 16 + e;
            af.u[e] = f2bf(xb[(kb + K) * N_POS + posBase + n]);
        }
        FragBF bfr;                                  // B[k][o=n], 16 consecutive c
        const float* wrow = w + (colBase + n) * C_DIM + kb + half * 16;
        #pragma unroll
        for (int e = 0; e < 16; ++e) bfr.u[e] = f2bf(wrow[e]);

        acc = __builtin_amdgcn_wmma_f32_16x16x32_bf16(false, af.v, false, bfr.v,
                                                      (short)0, acc, false, false);
    }
    const float bsv = bias[colBase + n];
    #pragma unroll
    for (int v = 0; v < 8; ++v) {
        int i = posBase + v + 8 * half;
        if (storeNO) out[(b * N_POS + i) * O + colBase + n] = f2bf(acc[v] + bsv);
        else         out[(b * O + colBase + n) * N_POS + i] = f2bf(acc[v] + bsv);
    }
}

// ---------------------------------------------------------------------------
// Fused flash attention + gated epilogue.
// Block = 8 waves = 128 query rows; each wave: 16 rows x 256 channels.
// K/V tiles double-buffered in LDS via async global->LDS loads.
// ---------------------------------------------------------------------------
__global__ void __launch_bounds__(256) attn_kernel(
    const unsigned short* __restrict__ QT,   // (B, N, 32) bf16
    const unsigned short* __restrict__ KT,   // (B, N, 32) bf16
    const unsigned short* __restrict__ Vf,   // (B, 256, N) bf16
    const float* __restrict__ a,             // (B, 256, N)
    const float* __restrict__ cmap,          // (B, 1, 128, 128)
    const float* __restrict__ gamma,
    float* __restrict__ outp)                // (B, 256, N)
{
    __shared__ __align__(16) unsigned short kbuf[2][JT * QK];      // 2 x 4 KB
    __shared__ __align__(16) unsigned short vbuf[2][C_DIM * JT];   // 2 x 32 KB
    __shared__ __align__(16) unsigned short ldsP[8][16 * JT];      // 8 x 2 KB

    const int lane = threadIdx.x & 31;
    const int w    = threadIdx.x >> 5;
    const int n    = lane & 15;
    const int half = lane >> 4;
    const int b    = blockIdx.y;
    const int rowBase = blockIdx.x * 128 + w * 16;

    unsigned short* myP = &ldsP[w][0];
    const unsigned short* Kb = KT + b * N_POS * QK;
    const unsigned short* Vb = Vf + b * C_DIM * N_POS;

    // Q A-fragment (held for the whole pass)
    FragBF qf;
    {
        const unsigned short* qr = QT + (b * N_POS + rowBase + n) * QK;
        qf.q[0] = *(const uint4*)(qr + half * 16);
        qf.q[1] = *(const uint4*)(qr + half * 16 + 8);
    }
    float cf[8];
    #pragma unroll
    for (int v = 0; v < 8; ++v) {
        int r = rowBase + v + 8 * half;
        cf[v] = cmap[b * 16384 + (r >> 6) * 256 + (r & 63) * 2];
    }

    float m_i[8], l_i[8];
    v8f acc[16];
    #pragma unroll
    for (int v = 0; v < 8; ++v) { m_i[v] = -3.0e38f; l_i[v] = 0.0f; }
    #pragma unroll
    for (int cs = 0; cs < 16; ++cs) { v8f z = {}; acc[cs] = z; }

    // cooperative tile fill: K 64x32 (1 b128/lane), V 256x64 (8 b128/lane)
    auto issue_fill = [&](int bi, int jBase) {
        {
            int j  = w * 8 + (lane >> 2);
            int ck = (lane & 3) * 8;
            cp16_g2l(&kbuf[bi][j * QK + ck], Kb + (jBase + j) * QK + ck);
        }
        #pragma unroll
        for (int i = 0; i < 8; ++i) {
            int cch = w * 32 + i * 4 + (lane >> 3);
            int k   = (lane & 7) * 8;
            cp16_g2l(&vbuf[bi][cch * JT + k], Vb + cch * N_POS + jBase + k);
        }
    };

    issue_fill(0, 0);

    for (int t = 0; t < NTILE; ++t) {
        wait_fill();
        __syncthreads();                         // tile t visible to all waves
        const int cur = t & 1;
        if (t + 1 < NTILE) issue_fill(cur ^ 1, (t + 1) * JT);

        // ---- S = Q K^T over 4 column subtiles ----
        FragBF kf;
        v8f z = {};
        v8f s0, s1, s2, s3;
        const unsigned short* kb0 = &kbuf[cur][0];
        kf.q[0] = *(const uint4*)(kb0 + (0 * 16 + n) * QK + half * 16);
        kf.q[1] = *(const uint4*)(kb0 + (0 * 16 + n) * QK + half * 16 + 8);
        s0 = __builtin_amdgcn_wmma_f32_16x16x32_bf16(false, qf.v, false, kf.v, (short)0, z, false, false);
        kf.q[0] = *(const uint4*)(kb0 + (1 * 16 + n) * QK + half * 16);
        kf.q[1] = *(const uint4*)(kb0 + (1 * 16 + n) * QK + half * 16 + 8);
        s1 = __builtin_amdgcn_wmma_f32_16x16x32_bf16(false, qf.v, false, kf.v, (short)0, z, false, false);
        kf.q[0] = *(const uint4*)(kb0 + (2 * 16 + n) * QK + half * 16);
        kf.q[1] = *(const uint4*)(kb0 + (2 * 16 + n) * QK + half * 16 + 8);
        s2 = __builtin_amdgcn_wmma_f32_16x16x32_bf16(false, qf.v, false, kf.v, (short)0, z, false, false);
        kf.q[0] = *(const uint4*)(kb0 + (3 * 16 + n) * QK + half * 16);
        kf.q[1] = *(const uint4*)(kb0 + (3 * 16 + n) * QK + half * 16 + 8);
        s3 = __builtin_amdgcn_wmma_f32_16x16x32_bf16(false, qf.v, false, kf.v, (short)0, z, false, false);

        // ---- online softmax (per-row scale cd[i]) ----
        float p[4][8], alpha[8];
        #pragma unroll
        for (int v = 0; v < 8; ++v) {
            float l0 = s0[v] * cf[v], l1 = s1[v] * cf[v];
            float l2 = s2[v] * cf[v], l3 = s3[v] * cf[v];
            float mx = fmaxf(fmaxf(l0, l1), fmaxf(l2, l3));
            mx = fmaxf(mx, __shfl_xor(mx, 1, 32));
            mx = fmaxf(mx, __shfl_xor(mx, 2, 32));
            mx = fmaxf(mx, __shfl_xor(mx, 4, 32));
            mx = fmaxf(mx, __shfl_xor(mx, 8, 32));
            float mnew = fmaxf(m_i[v], mx);
            alpha[v] = __expf(m_i[v] - mnew);
            p[0][v] = __expf(l0 - mnew); p[1][v] = __expf(l1 - mnew);
            p[2][v] = __expf(l2 - mnew); p[3][v] = __expf(l3 - mnew);
            float rs = (p[0][v] + p[1][v]) + (p[2][v] + p[3][v]);
            rs += __shfl_xor(rs, 1, 32);
            rs += __shfl_xor(rs, 2, 32);
            rs += __shfl_xor(rs, 4, 32);
            rs += __shfl_xor(rs, 8, 32);
            l_i[v] = l_i[v] * alpha[v] + rs;
            m_i[v] = mnew;
        }
        #pragma unroll
        for (int cs = 0; cs < 16; ++cs) {
            #pragma unroll
            for (int v = 0; v < 8; ++v) acc[cs][v] *= alpha[v];
        }

        // ---- P: C/D layout -> A layout through per-wave LDS ----
        #pragma unroll
        for (int v = 0; v < 8; ++v) {
            int row = v + 8 * half;
            myP[row * JT + n]      = f2bf(p[0][v]);
            myP[row * JT + 16 + n] = f2bf(p[1][v]);
            myP[row * JT + 32 + n] = f2bf(p[2][v]);
            myP[row * JT + 48 + n] = f2bf(p[3][v]);
        }
        __syncthreads();                         // store->load fence (and TBAA)
        FragBF pf0, pf1;
        pf0.q[0] = *(const uint4*)(myP + n * JT + half * 16);
        pf0.q[1] = *(const uint4*)(myP + n * JT + half * 16 + 8);
        pf1.q[0] = *(const uint4*)(myP + n * JT + 32 + half * 16);
        pf1.q[1] = *(const uint4*)(myP + n * JT + 32 + half * 16 + 8);

        // ---- O += P @ V over all 256 channels ----
        #pragma unroll
        for (int cs = 0; cs < 16; ++cs) {
            const unsigned short* vr = &vbuf[cur][(cs * 16 + n) * JT];
            FragBF va, vb2;
            va.q[0]  = *(const uint4*)(vr + half * 16);
            va.q[1]  = *(const uint4*)(vr + half * 16 + 8);
            vb2.q[0] = *(const uint4*)(vr + 32 + half * 16);
            vb2.q[1] = *(const uint4*)(vr + 32 + half * 16 + 8);
            acc[cs] = __builtin_amdgcn_wmma_f32_16x16x32_bf16(false, pf0.v, false, va.v,
                                                              (short)0, acc[cs], false, false);
            acc[cs] = __builtin_amdgcn_wmma_f32_16x16x32_bf16(false, pf1.v, false, vb2.v,
                                                              (short)0, acc[cs], false, false);
        }
    }

    // ---------------- epilogue: out = gamma*a*cd + (1-cd) * O/l ----------------
    float linv[8];
    #pragma unroll
    for (int v = 0; v < 8; ++v) linv[v] = 1.0f / l_i[v];

    const float gm = gamma[0];
    const int   i_g = rowBase + n;
    const float cf_e  = cmap[b * 16384 + (i_g >> 6) * 256 + (i_g & 63) * 2];
    const float one_m = 1.0f - cf_e;
    float* myT = (float*)myP;                    // reuse per-wave region, 16x16 f32

    for (int cs = 0; cs < 16; ++cs) {
        __syncthreads();
        AccF st;
        #pragma unroll
        for (int v = 0; v < 8; ++v) st.f[v] = acc[cs][v] * linv[v];
        *(float4*)(myT + n * 16 + 8 * half)     = st.q[0];   // tile[ch_l=n][i_l]
        *(float4*)(myT + n * 16 + 8 * half + 4) = st.q[1];
        __syncthreads();
        #pragma unroll
        for (int r = 0; r < 8; ++r) {
            int ch_l = 2 * r + half;
            float oval = myT[ch_l * 16 + n];
            int ch = cs * 16 + ch_l;
            int gidx = (b * C_DIM + ch) * N_POS + i_g;
            float av = a[gidx];
            outp[gidx] = gm * av * cf_e + one_m * oval;
        }
    }
}

// ---------------------------------------------------------------------------
extern "C" void kernel_launch(void* const* d_in, const int* in_sizes, int n_in,
                              void* d_out, int out_size, void* d_ws, size_t ws_size,
                              hipStream_t stream) {
    const float* a  = (const float*)d_in[0];
    const float* bI = (const float*)d_in[1];
    const float* c  = (const float*)d_in[2];
    const float* wq = (const float*)d_in[3];
    const float* bq = (const float*)d_in[4];
    const float* wk = (const float*)d_in[5];
    const float* bk = (const float*)d_in[6];
    const float* wv = (const float*)d_in[7];
    const float* bv = (const float*)d_in[8];
    const float* gamma = (const float*)d_in[9];

    unsigned short* QT = (unsigned short*)d_ws;                 // 1 MB
    unsigned short* KT = QT + (size_t)B_SZ * N_POS * QK;        // 1 MB
    unsigned short* Vf = KT + (size_t)B_SZ * N_POS * QK;        // 8 MB

    proj_kernel<<<dim3(N_POS / 64, QK / 16, B_SZ), 128, 0, stream>>>(a,  wq, bq, QT, QK, 1);
    proj_kernel<<<dim3(N_POS / 64, QK / 16, B_SZ), 128, 0, stream>>>(bI, wk, bk, KT, QK, 1);
    proj_kernel<<<dim3(N_POS / 64, C_DIM / 16, B_SZ), 128, 0, stream>>>(a, wv, bv, Vf, C_DIM, 0);

    attn_kernel<<<dim3(N_POS / 128, B_SZ), 256, 0, stream>>>(QT, KT, Vf, a, c, gamma, (float*)d_out);
}
